// TransformerBlockRef_55542517072105
// MI455X (gfx1250) — matrix operand order repaired and checked by reading, not compiled, not run on hardware
//
#include <hip/hip_runtime.h>
#include <hip/hip_bf16.h>
#include <math.h>

// ---------------------------------------------------------------------------
// Transformer block (pre-norm attn + FFN) for MI455X / gfx1250.
// Compute-bound (~240 GFLOP vs ~100MB HBM traffic) -> bf16 WMMA with f32 acc.
// GEMMs use double-buffered LDS tiles filled with GLOBAL_LOAD_ASYNC_TO_LDS
// (ASYNCcnt path, no VGPR bounce), one barrier per K-step.
// ---------------------------------------------------------------------------

#define D_MODEL 1024
#define D_FF    4096
#define SEQ     2048
#define BATCH   2
#define NHEAD   16
#define DKH     64
#define MROWS   (BATCH * SEQ)     // 4096 token rows

typedef __attribute__((ext_vector_type(16))) __bf16 v16bf;
typedef __attribute__((ext_vector_type(8)))  __bf16 v8bf;
typedef __attribute__((ext_vector_type(8)))  float  v8f;

static __device__ __forceinline__ v8f wmma_bf16(v16bf a, v16bf b, v8f c) {
  // v_wmma_f32_16x16x32_bf16  D = A*B + C
  return __builtin_amdgcn_wmma_f32_16x16x32_bf16(
      /*neg_a=*/false, a, /*neg_b=*/false, b,
      /*c_mod=*/(short)0, c, /*reuse_a=*/false, /*reuse_b=*/false);
}

// A-fragment (16x32 bf16, row-major source with row stride `stride` halves).
// Contiguous k, matching B: lanes 0-15 hold M=lane, K=0..15;
//             lanes 16-31 hold M=lane-16, K=16..31.
static __device__ __forceinline__ v16bf load_frag_a(const __bf16* p0, int stride, int lane) {
  const int r  = lane & 15;
  const int kh = (lane >> 4) << 4;            // 0 or 16
  const __bf16* p = p0 + (size_t)r * stride + kh;
  v8bf lo = *(const v8bf*)(p);
  v8bf hi = *(const v8bf*)(p + 8);
  v16bf v;
#pragma unroll
  for (int i = 0; i < 8; ++i) { v[i] = lo[i]; v[i + 8] = hi[i]; }
  return v;
}

// B-fragment (32x16 bf16). Source stored as [N][K] row-major, stride halves.
// Contiguous k, matching A: lanes 0-15 hold N=lane, K=0..15; lanes 16-31 hold N=lane-16, K=16..31.
static __device__ __forceinline__ v16bf load_frag_b(const __bf16* p0, int stride, int lane) {
  const int n  = lane & 15;
  const int k0 = (lane >> 4) << 4;            // 0 or 16
  const __bf16* p = p0 + (size_t)n * stride + k0;
  v8bf lo = *(const v8bf*)(p);
  v8bf hi = *(const v8bf*)(p + 8);
  v16bf v;
#pragma unroll
  for (int i = 0; i < 8; ++i) { v[i] = lo[i]; v[i + 8] = hi[i]; }
  return v;
}

// ---------------------------------------------------------------------------
// LayerNorm (fp32 in) -> bf16 out.  One 256-thread block per token row.
// ---------------------------------------------------------------------------
__global__ __launch_bounds__(256)
void ln_bf16_kernel(const float* __restrict__ x, const float* __restrict__ g,
                    const float* __restrict__ be, __bf16* __restrict__ out) {
  __shared__ float red[8];
  __shared__ float stat;
  const int row = blockIdx.x;
  const int t   = threadIdx.x;
  const float* xr = x + (size_t)row * D_MODEL;

  float v[4];
  float s = 0.f;
#pragma unroll
  for (int i = 0; i < 4; ++i) { v[i] = xr[t + i * 256]; s += v[i]; }
#pragma unroll
  for (int m = 1; m < 32; m <<= 1) s += __shfl_xor(s, m, 32);
  if ((t & 31) == 0) red[t >> 5] = s;
  __syncthreads();
  if (t == 0) { float a = 0.f; for (int i = 0; i < 8; ++i) a += red[i]; stat = a * (1.f / D_MODEL); }
  __syncthreads();
  const float mu = stat;

  float s2 = 0.f;
#pragma unroll
  for (int i = 0; i < 4; ++i) { float d = v[i] - mu; s2 += d * d; }
#pragma unroll
  for (int m = 1; m < 32; m <<= 1) s2 += __shfl_xor(s2, m, 32);
  __syncthreads();
  if ((t & 31) == 0) red[t >> 5] = s2;
  __syncthreads();
  if (t == 0) { float a = 0.f; for (int i = 0; i < 8; ++i) a += red[i];
                stat = rsqrtf(a * (1.f / D_MODEL) + 1e-5f); }
  __syncthreads();
  const float rs = stat;

#pragma unroll
  for (int i = 0; i < 4; ++i) {
    const int c = t + i * 256;
    out[(size_t)row * D_MODEL + c] = (__bf16)((v[i] - mu) * rs * g[c] + be[c]);
  }
}

// ---------------------------------------------------------------------------
// fp32 -> bf16 elementwise (weights)
// ---------------------------------------------------------------------------
__global__ void cvt_bf16_kernel(const float* __restrict__ in, __bf16* __restrict__ out, int n) {
  int i = blockIdx.x * blockDim.x + threadIdx.x;
  const int stride = gridDim.x * blockDim.x;
  for (; i < n; i += stride) out[i] = (__bf16)in[i];
}

// ---------------------------------------------------------------------------
// GEMM: out[M,N] = A[M,K] @ W[N,K]^T + bias  (+ epilogue)
//   128x128 block tile, K-tile 32, 8 waves, each wave -> 64x32 output.
//   Double-buffered LDS staged with GLOBAL_LOAD_ASYNC_TO_LDS_B128.
//   flags bit0: exact GELU.
// ---------------------------------------------------------------------------
#define GEMM_GELU 1

__global__ __launch_bounds__(256)
void gemm_bf16_kernel(const __bf16* __restrict__ A, const __bf16* __restrict__ Bw,
                      const float* __restrict__ bias, const float* __restrict__ resid,
                      float* __restrict__ outF, __bf16* __restrict__ outB,
                      int N, int K, int flags) {
  __shared__ __bf16 lA[2][128 * 32];
  __shared__ __bf16 lB[2][128 * 32];

  const int t    = threadIdx.x;
  const int lane = t & 31;
  const int w    = t >> 5;
  const int wm   = (w & 1) * 64;     // wave M offset in block tile
  const int wn   = (w >> 1) * 32;    // wave N offset in block tile
  const int m0   = blockIdx.y * 128;
  const int n0   = blockIdx.x * 128;

  // Each thread owns two 8-half segments of the 128x32 A and B tiles.
  // Async copy: per-lane LDS dest addr (= low 32 bits of generic pointer,
  // per ISA flat-aperture rule) in a VGPR, global addr in a VGPR pair.
  auto stage = [&](int k0, int buf) {
#pragma unroll
    for (int i = 0; i < 2; ++i) {
      const int seg = t + (i << 8);
      const int r   = seg >> 2;
      const int c   = (seg & 3) << 3;
      const void* ga = &A [(size_t)(m0 + r) * K + k0 + c];
      const void* gb = &Bw[(size_t)(n0 + r) * K + k0 + c];
      const unsigned la = (unsigned)(size_t)&lA[buf][r * 32 + c];
      const unsigned lb = (unsigned)(size_t)&lB[buf][r * 32 + c];
      asm volatile(
          "global_load_async_to_lds_b128 %0, %2, off\n\t"
          "global_load_async_to_lds_b128 %1, %3, off"
          :
          : "v"(la), "v"(lb), "v"(ga), "v"(gb)
          : "memory");
    }
  };

  const v8f zero = {};
  v8f acc[4][2];
#pragma unroll
  for (int i = 0; i < 4; ++i)
#pragma unroll
    for (int j = 0; j < 2; ++j) acc[i][j] = zero;

  stage(0, 0);                          // prologue: tile 0 in flight

  const int nk = K >> 5;
  for (int kt = 0; kt < nk; ++kt) {
    const int buf = kt & 1;
    asm volatile("s_wait_asynccnt 0x0" ::: "memory");  // my async copies landed
    __syncthreads();                                   // everyone's landed

    if (kt + 1 < nk) stage((kt + 1) << 5, buf ^ 1);    // overlap next tile

    v16bf af[4], bfr[2];
#pragma unroll
    for (int i = 0; i < 4; ++i) af[i]  = load_frag_a(&lA[buf][(wm + i * 16) * 32], 32, lane);
#pragma unroll
    for (int j = 0; j < 2; ++j) bfr[j] = load_frag_b(&lB[buf][(wn + j * 16) * 32], 32, lane);
#pragma unroll
    for (int i = 0; i < 4; ++i)
#pragma unroll
      for (int j = 0; j < 2; ++j) acc[i][j] = wmma_bf16(af[i], bfr[j], acc[i][j]);
    // No trailing barrier: ds reads are consumed (s_wait_dscnt) before the
    // WMMAs; this buffer is only re-written after the NEXT iteration's
    // s_wait_asynccnt + barrier, which orders against those reads.
  }

  // Epilogue. C layout: vgpr e -> row e (+8 for lanes 16-31), col = lane&15.
  const int cn = lane & 15;
  const int rb = (lane >> 4) << 3;
#pragma unroll
  for (int i = 0; i < 4; ++i) {
#pragma unroll
    for (int j = 0; j < 2; ++j) {
      const int gn = n0 + wn + j * 16 + cn;
      const float bv = bias ? bias[gn] : 0.f;
#pragma unroll
      for (int e = 0; e < 8; ++e) {
        const int gm = m0 + wm + i * 16 + rb + e;
        float val = acc[i][j][e] + bv;
        if (flags & GEMM_GELU)
          val = 0.5f * val * (1.f + erff(val * 0.70710678118654752f));
        const size_t off = (size_t)gm * N + gn;
        if (resid) val += resid[off];
        if (outF)  outF[off] = val;
        if (outB)  outB[off] = (__bf16)val;
      }
    }
  }
}

// ---------------------------------------------------------------------------
// Flash attention: one (b, h, 64 q-rows) tile per 128-thread block (4 waves).
// q/k/v/ctx are bf16 [B, S, H, DKH]; row stride = H*DKH = 1024 halves.
// Per 32-key chunk: 4 WMMAs QK^T + online softmax + 4 WMMAs P.V.
// ---------------------------------------------------------------------------
__global__ __launch_bounds__(128)
void attn_kernel(const __bf16* __restrict__ q, const __bf16* __restrict__ k,
                 const __bf16* __restrict__ v, __bf16* __restrict__ ctx) {
  __shared__ __bf16 lp[4][16 * 32];       // per-wave P tile (16 q-rows x 32 keys)

  const int t    = threadIdx.x;
  const int lane = t & 31;
  const int w    = t >> 5;
  const int cn   = lane & 15;
  const int rb   = (lane >> 4) << 3;
  const int b    = blockIdx.z;
  const int h    = blockIdx.y;
  const int q0   = blockIdx.x * 64 + w * 16;

  const size_t base = (size_t)b * SEQ * D_MODEL + (size_t)h * DKH;
  const __bf16* qb = q + base + (size_t)q0 * D_MODEL;
  const __bf16* kb = k + base;
  const __bf16* vb = v + base;
  __bf16*       cb = ctx + base + (size_t)q0 * D_MODEL;

  // Q fragments: d = 0..31 and 32..63
  const v16bf qf0 = load_frag_a(qb,      D_MODEL, lane);
  const v16bf qf1 = load_frag_a(qb + 32, D_MODEL, lane);

  const v8f zero = {};
  v8f acc[4];
#pragma unroll
  for (int d = 0; d < 4; ++d) acc[d] = zero;
  float mrow[8], lrow[8];
#pragma unroll
  for (int e = 0; e < 8; ++e) { mrow[e] = -1e30f; lrow[e] = 0.f; }

  const float scale = 0.125f;             // 1/sqrt(64)

  for (int kk0 = 0; kk0 < SEQ; kk0 += 32) {
    // scores for keys [kk0, kk0+16) and [kk0+16, kk0+32)
    v8f s0 = zero, s1 = zero;
    {
      v16bf kf;
      kf = load_frag_b(kb + (size_t)kk0 * D_MODEL,            D_MODEL, lane);
      s0 = wmma_bf16(qf0, kf, s0);
      kf = load_frag_b(kb + (size_t)kk0 * D_MODEL + 32,       D_MODEL, lane);
      s0 = wmma_bf16(qf1, kf, s0);
      kf = load_frag_b(kb + (size_t)(kk0 + 16) * D_MODEL,     D_MODEL, lane);
      s1 = wmma_bf16(qf0, kf, s1);
      kf = load_frag_b(kb + (size_t)(kk0 + 16) * D_MODEL + 32, D_MODEL, lane);
      s1 = wmma_bf16(qf1, kf, s1);
    }

    // online softmax; rows live across 16-lane half-groups
    __bf16* lpw = lp[w];
    float alpha[8];
#pragma unroll
    for (int e = 0; e < 8; ++e) {
      const float a0 = s0[e] * scale;
      const float a1 = s1[e] * scale;
      float rmax = fmaxf(a0, a1);
#pragma unroll
      for (int m = 1; m < 16; m <<= 1) rmax = fmaxf(rmax, __shfl_xor(rmax, m, 32));
      const float nm = fmaxf(mrow[e], rmax);
      const float al = __expf(mrow[e] - nm);
      const float p0 = __expf(a0 - nm);
      const float p1 = __expf(a1 - nm);
      float rs = p0 + p1;
#pragma unroll
      for (int m = 1; m < 16; m <<= 1) rs += __shfl_xor(rs, m, 32);
      lrow[e]  = lrow[e] * al + rs;
      mrow[e]  = nm;
      alpha[e] = al;
      lpw[(rb + e) * 32 + cn]      = (__bf16)p0;
      lpw[(rb + e) * 32 + 16 + cn] = (__bf16)p1;
    }
#pragma unroll
    for (int d = 0; d < 4; ++d)
#pragma unroll
      for (int e = 0; e < 8; ++e) acc[d][e] *= alpha[e];

    __syncthreads();                         // P tile visible (uniform trip count)
    const v16bf pa = load_frag_a(lpw, 32, lane);

    // ctx += P(16x32) @ V(32x64), four 16-wide d tiles
#pragma unroll
    for (int d = 0; d < 4; ++d) {
      v16bf vf;
      const int ks = rb << 1;                // (lane>>4)*16 key offset
      const __bf16* vp = vb + (size_t)(kk0 + ks) * D_MODEL + d * 16 + cn;
#pragma unroll
      for (int u = 0; u < 16; ++u) vf[u] = vp[(size_t)u * D_MODEL];
      acc[d] = wmma_bf16(pa, vf, acc[d]);
    }
    __syncthreads();                         // protect lp before next chunk
  }

#pragma unroll
  for (int d = 0; d < 4; ++d)
#pragma unroll
    for (int e = 0; e < 8; ++e)
      cb[(size_t)(rb + e) * D_MODEL + d * 16 + cn] = (__bf16)(acc[d][e] / lrow[e]);
}

// ---------------------------------------------------------------------------
// Host-side orchestration
// ---------------------------------------------------------------------------
extern "C" void kernel_launch(void* const* d_in, const int* in_sizes, int n_in,
                              void* d_out, int out_size, void* d_ws, size_t ws_size,
                              hipStream_t stream) {
  (void)in_sizes; (void)n_in; (void)out_size; (void)ws_size;

  const float* x   = (const float*)d_in[0];
  const float* wq  = (const float*)d_in[1];  const float* bq = (const float*)d_in[2];
  const float* wk  = (const float*)d_in[3];  const float* bk = (const float*)d_in[4];
  const float* wv  = (const float*)d_in[5];  const float* bv = (const float*)d_in[6];
  const float* wo  = (const float*)d_in[7];  const float* bo = (const float*)d_in[8];
  const float* w1  = (const float*)d_in[9];  const float* b1 = (const float*)d_in[10];
  const float* w2  = (const float*)d_in[11]; const float* b2 = (const float*)d_in[12];
  const float* g1  = (const float*)d_in[13]; const float* be1 = (const float*)d_in[14];
  const float* g2  = (const float*)d_in[15]; const float* be2 = (const float*)d_in[16];
  float* out = (float*)d_out;

  // ---- workspace carve-up (all sizes 128B-aligned by construction) ----
  char* ws = (char*)d_ws;
  __bf16* h1  = (__bf16*)ws;  ws += (size_t)MROWS * D_MODEL * 2;  // LN1(x)
  __bf16* wqb = (__bf16*)ws;  ws += (size_t)D_MODEL * D_MODEL * 2;
  __bf16* wkb = (__bf16*)ws;  ws += (size_t)D_MODEL * D_MODEL * 2;
  __bf16* wvb = (__bf16*)ws;  ws += (size_t)D_MODEL * D_MODEL * 2;
  __bf16* wob = (__bf16*)ws;  ws += (size_t)D_MODEL * D_MODEL * 2;
  __bf16* w1b = (__bf16*)ws;  ws += (size_t)D_FF   * D_MODEL * 2;
  __bf16* w2b = (__bf16*)ws;  ws += (size_t)D_MODEL * D_FF   * 2;
  __bf16* qb  = (__bf16*)ws;  ws += (size_t)MROWS * D_MODEL * 2;
  __bf16* kb  = (__bf16*)ws;  ws += (size_t)MROWS * D_MODEL * 2;
  __bf16* vb  = (__bf16*)ws;  ws += (size_t)MROWS * D_MODEL * 2;
  __bf16* ctx = (__bf16*)ws;  ws += (size_t)MROWS * D_MODEL * 2;
  float*  x1  = (float*)ws;   ws += (size_t)MROWS * D_MODEL * 4;  // x + attn out
  __bf16* h2  = (__bf16*)ws;  ws += (size_t)MROWS * D_MODEL * 2;  // LN2(x1)
  __bf16* f1  = (__bf16*)ws;  ws += (size_t)MROWS * D_FF * 2;     // GELU(h2@w1^T+b1)

  const dim3 blk256(256);
  const dim3 gD(D_MODEL / 128, MROWS / 128);   // N=1024 GEMMs
  const dim3 gF(D_FF    / 128, MROWS / 128);   // N=4096 GEMM

  // 1) LN1 -> bf16
  ln_bf16_kernel<<<MROWS, blk256, 0, stream>>>(x, g1, be1, h1);

  // 2) weights -> bf16
  cvt_bf16_kernel<<<2048, blk256, 0, stream>>>(wq, wqb, D_MODEL * D_MODEL);
  cvt_bf16_kernel<<<2048, blk256, 0, stream>>>(wk, wkb, D_MODEL * D_MODEL);
  cvt_bf16_kernel<<<2048, blk256, 0, stream>>>(wv, wvb, D_MODEL * D_MODEL);
  cvt_bf16_kernel<<<2048, blk256, 0, stream>>>(wo, wob, D_MODEL * D_MODEL);
  cvt_bf16_kernel<<<4096, blk256, 0, stream>>>(w1, w1b, D_FF * D_MODEL);
  cvt_bf16_kernel<<<4096, blk256, 0, stream>>>(w2, w2b, D_MODEL * D_FF);

  // 3) Q/K/V projections
  gemm_bf16_kernel<<<gD, blk256, 0, stream>>>(h1, wqb, bq, nullptr, nullptr, qb,
                                              D_MODEL, D_MODEL, 0);
  gemm_bf16_kernel<<<gD, blk256, 0, stream>>>(h1, wkb, bk, nullptr, nullptr, kb,
                                              D_MODEL, D_MODEL, 0);
  gemm_bf16_kernel<<<gD, blk256, 0, stream>>>(h1, wvb, bv, nullptr, nullptr, vb,
                                              D_MODEL, D_MODEL, 0);

  // 4) flash attention
  attn_kernel<<<dim3(SEQ / 64, NHEAD, BATCH), dim3(128), 0, stream>>>(qb, kb, vb, ctx);

  // 5) out projection + residual -> x1 (f32)
  gemm_bf16_kernel<<<gD, blk256, 0, stream>>>(ctx, wob, bo, x, x1, nullptr,
                                              D_MODEL, D_MODEL, 0);

  // 6) LN2 -> bf16
  ln_bf16_kernel<<<MROWS, blk256, 0, stream>>>(x1, g2, be2, h2);

  // 7) FFN1 with exact GELU -> bf16
  gemm_bf16_kernel<<<gF, blk256, 0, stream>>>(h2, w1b, b1, nullptr, nullptr, f1,
                                              D_FF, D_MODEL, GEMM_GELU);

  // 8) FFN2 + residual -> d_out (f32)
  gemm_bf16_kernel<<<gD, blk256, 0, stream>>>(f1, w2b, b2, x1, out, nullptr,
                                              D_MODEL, D_FF, 0);
}
